// EpisodicMemory_43224550866999
// MI455X (gfx1250) — matrix operand order repaired and checked by reading, not compiled, not run on hardware
//
#include <hip/hip_runtime.h>
#include <hip/hip_bf16.h>
#include <math.h>

typedef __attribute__((ext_vector_type(16))) __bf16 v16bf;
typedef __attribute__((ext_vector_type(8)))  float  v8f;

#define BETA 2.0f
#define AROW 264          // attn LDS row stride in shorts (16B-aligned rows)
#define AROW32 132        // same, in dwords (bank advance 4/lane -> conflict-free b128)

union Frag16 { v16bf v; unsigned int d[8]; uint4 q[2]; };

// Exact RNE (cold path: values pre-pack only)
__device__ __forceinline__ unsigned int f2bf_rne(float f) {
    union { float f; unsigned int u; } c; c.f = f;
    unsigned int u = c.u;
    return (u + 0x7FFFu + ((u >> 16) & 1u)) >> 16;
}
__device__ __forceinline__ unsigned int pack_bf16_rne(float a, float b) {
    return (f2bf_rne(a) & 0xFFFFu) | (f2bf_rne(b) << 16);
}

// Hot path: round-half-up (<=0.5 ULP), 2x v_add + 1x v_perm_b32 per pair
__device__ __forceinline__ unsigned int bfr(float f) {
    union { float f; unsigned int u; } c; c.f = f;
    return c.u + 0x8000u;
}
__device__ __forceinline__ unsigned int pack_bf16(float a, float b) {
    // take bytes 3:2 of (a+rnd) as low half, bytes 3:2 of (b+rnd) as high half
    return __builtin_amdgcn_perm(bfr(b), bfr(a), 0x07060302u);
}
__device__ __forceinline__ unsigned short f2bf(float f) {
    return (unsigned short)(bfr(f) >> 16);    // folds into d16-hi store
}

// ---------------------------------------------------------------------------
// Pre-pass: values f32[240][256] -> K-pair packed bf16 dwords, bank-swizzled:
// vpk[kk*256 + (n ^ (((kk>>3)&1)<<4))] = pack(values[2kk][n], values[2kk+1][n])
// ---------------------------------------------------------------------------
__global__ void pack_values_kernel(const float* __restrict__ values,
                                   unsigned int* __restrict__ vpk) {
    int idx = blockIdx.x * blockDim.x + threadIdx.x;   // 0..32767
    int kk = idx >> 8;
    int n  = idx & 255;
    float a = (2 * kk     < 240) ? values[(2 * kk)     * 256 + n] : 0.0f;
    float b = (2 * kk + 1 < 240) ? values[(2 * kk + 1) * 256 + n] : 0.0f;
    int sw = ((kk >> 3) & 1) << 4;
    vpk[kk * 256 + (n ^ sw)] = pack_bf16_rne(a, b);
}

// ---------------------------------------------------------------------------
// Main kernel: 256 threads = 8 waves; wave w owns 16 tokens (one M-tile).
// ---------------------------------------------------------------------------
__launch_bounds__(256)
__global__ void episodic_kernel(const float* __restrict__ q,
                                const float* __restrict__ W1,
                                const float* __restrict__ b1,
                                const float* __restrict__ W2,
                                const float* __restrict__ b2,
                                const float* __restrict__ keys,
                                const unsigned int* __restrict__ vpk,
                                float* __restrict__ out_content,
                                float* __restrict__ out_attn) {
    __shared__ __align__(16) unsigned int   vals_lds[128 * 256];   // 128 KB packed bf16 values
    __shared__ __align__(16) unsigned short attn_lds[128 * AROW];  // 66 KB bf16 attn (padded)

    const int tid  = threadIdx.x;
    const int wave = tid >> 5;          // 0..7  == local M-tile
    const int lane = tid & 31;
    const int hh   = lane >> 4;         // half-wave id
    const int l16  = lane & 15;
    const int mtb  = wave * 16;         // local row base of this wave's tile
    const float hsel = (hh == 0) ? 1.0f : 0.0f;   // branchless lane select

    // ---- kick off async copy: packed values (128 KB) -> LDS, overlapped ----
    {
        unsigned lds_base = (unsigned)(uintptr_t)(&vals_lds[0]);
        #pragma unroll
        for (int it = 0; it < 32; ++it) {
            unsigned off  = (unsigned)((it * 256 + tid) * 16);   // byte offset
            unsigned ldsa = lds_base + off;
            asm volatile("global_load_async_to_lds_b128 %0, %1, %2"
                         :: "v"(ldsa), "v"(off), "s"(vpk)
                         : "memory");
        }
    }

    const long long tok_base = (long long)blockIdx.x * 128 + mtb;
    const long long token    = tok_base + l16;         // A-matrix row token

    // ---- front end: x -> GELU MLP -> q8 (unit norm), one token per lane ----
    float x[14];
    #pragma unroll
    for (int k = 0; k < 14; ++k) x[k] = q[token * 14 + k];

    float h[32];
    #pragma unroll
    for (int j = 0; j < 32; ++j) {
        float acc = b1[j];
        #pragma unroll
        for (int k = 0; k < 14; ++k) acc += x[k] * W1[j * 14 + k];
        h[j] = 0.5f * acc * (1.0f + erff(acc * 0.70710678118f));  // exact GELU
    }
    float q8[8];
    float nrm = 0.0f;
    #pragma unroll
    for (int j = 0; j < 8; ++j) {
        float acc = b2[j];
        #pragma unroll
        for (int k = 0; k < 32; ++k) acc += h[k] * W2[j * 32 + k];
        q8[j] = acc;
        nrm  += acc * acc;
    }
    nrm = 1.0f / fmaxf(sqrtf(nrm), 1e-12f);
    #pragma unroll
    for (int j = 0; j < 8; ++j) q8[j] *= nrm * hsel;  // zero for half 1 (unused)

    // ---- sim = q8 (16x8 pad 32) x keys^T (8x240 pad) via bf16 WMMA ----
    Frag16 fa;
    #pragma unroll
    for (int i = 0; i < 8; ++i) fa.d[i] = 0u;
    #pragma unroll
    for (int i = 0; i < 4; ++i) fa.d[i] = pack_bf16(q8[2 * i], q8[2 * i + 1]);

    v8f sim[15];
    #pragma unroll
    for (int nt = 0; nt < 15; ++nt) {
        int n = nt * 16 + l16;                           // slot column
        const float2* kp = (const float2*)(keys + n * 8);
        Frag16 fb;
        #pragma unroll
        for (int i = 0; i < 8; ++i) fb.d[i] = 0u;
        #pragma unroll
        for (int i = 0; i < 4; ++i) {                    // unconditional loads
            float2 kv = kp[i];
            fb.d[i] = pack_bf16(kv.x * hsel, kv.y * hsel);
        }
        v8f c = {0.f, 0.f, 0.f, 0.f, 0.f, 0.f, 0.f, 0.f};
        sim[nt] = __builtin_amdgcn_wmma_f32_16x16x32_bf16(
            false, fa.v, false, fb.v, (short)0, c, false, false);
    }

    // ---- softmax over 240 slots, row-wise (row r+8*hh lives in VGPR r) ----
    float rmax[8], rinv[8];
    #pragma unroll
    for (int r = 0; r < 8; ++r) {
        float m = sim[0][r];
        #pragma unroll
        for (int nt = 1; nt < 15; ++nt) m = fmaxf(m, sim[nt][r]);
        m = fmaxf(m, __shfl_xor(m, 1, 32));
        m = fmaxf(m, __shfl_xor(m, 2, 32));
        m = fmaxf(m, __shfl_xor(m, 4, 32));
        m = fmaxf(m, __shfl_xor(m, 8, 32));
        rmax[r] = m;
    }
    #pragma unroll
    for (int r = 0; r < 8; ++r) {
        float s = 0.0f;
        #pragma unroll
        for (int nt = 0; nt < 15; ++nt) {
            float e = __expf(BETA * (sim[nt][r] - rmax[r]));
            sim[nt][r] = e;
            s += e;
        }
        s += __shfl_xor(s, 1, 32);
        s += __shfl_xor(s, 2, 32);
        s += __shfl_xor(s, 4, 32);
        s += __shfl_xor(s, 8, 32);
        rinv[r] = 1.0f / s;
    }

    // ---- write attn (f32 global, coalesced) + bf16 to LDS for big GEMM ----
    #pragma unroll
    for (int nt = 0; nt < 15; ++nt) {
        int n = nt * 16 + l16;
        #pragma unroll
        for (int r = 0; r < 8; ++r) {
            int m = r + 8 * hh;
            float a = sim[nt][r] * rinv[r];
            out_attn[(tok_base + m) * 240 + n] = a;
            attn_lds[(mtb + m) * AROW + n] = f2bf(a);
        }
    }
    {   // zero-pad slots 240..255
        int n = 240 + l16;
        #pragma unroll
        for (int r = 0; r < 8; ++r)
            attn_lds[(mtb + r + 8 * hh) * AROW + n] = 0;
    }

    // ---- complete async values copy, make LDS visible workgroup-wide ----
    asm volatile("s_wait_asynccnt 0" ::: "memory");
    __syncthreads();

    // ---- content = attn (16x256) x values (256x256) via bf16 WMMA ----
    const uint4* arow_q = (const uint4*)((const unsigned int*)attn_lds +
                                         (mtb + l16) * AROW32);   // 16B-aligned row
    const int csw = hh << 4;                                      // B column bank swizzle
    for (int pass = 0; pass < 2; ++pass) {
        v8f acc[8];
        #pragma unroll
        for (int j = 0; j < 8; ++j)
            acc[j] = (v8f){0.f, 0.f, 0.f, 0.f, 0.f, 0.f, 0.f, 0.f};

        for (int k0 = 0; k0 < 8; ++k0) {
            Frag16 fA;                                    // 2x ds_load_b128
            fA.q[0] = arow_q[k0 * 4 + 2 * hh];            // dwords k0*16 + 8*hh + 0..3
            fA.q[1] = arow_q[k0 * 4 + 2 * hh + 1];        // dwords k0*16 + 8*hh + 4..7
            #pragma unroll
            for (int j = 0; j < 8; ++j) {
                int nt = pass * 8 + j;
                int c  = (nt * 16 + l16) ^ csw;           // swizzled column
                Frag16 fB;
                #pragma unroll
                for (int i = 0; i < 8; ++i) {
                    int kk = k0 * 16 + 8 * hh + i;        // (k0*32 + 16*hh + 2i)/2
                    fB.d[i] = vals_lds[kk * 256 + c];
                }
                acc[j] = __builtin_amdgcn_wmma_f32_16x16x32_bf16(
                    false, fA.v, false, fB.v, (short)0, acc[j], false, false);
            }
        }
        #pragma unroll
        for (int j = 0; j < 8; ++j) {
            int n = (pass * 8 + j) * 16 + l16;
            #pragma unroll
            for (int r = 0; r < 8; ++r)
                out_content[(tok_base + r + 8 * hh) * 256 + n] = acc[j][r];
        }
    }
}

extern "C" void kernel_launch(void* const* d_in, const int* in_sizes, int n_in,
                              void* d_out, int out_size, void* d_ws, size_t ws_size,
                              hipStream_t stream) {
    const float* query  = (const float*)d_in[0];
    const float* W1     = (const float*)d_in[1];
    const float* b1     = (const float*)d_in[2];
    const float* W2     = (const float*)d_in[3];
    const float* b2     = (const float*)d_in[4];
    const float* values = (const float*)d_in[5];
    const float* keys   = (const float*)d_in[6];

    const long long N = (long long)in_sizes[0] / 14;   // 262144 tokens
    float* out_content = (float*)d_out;
    float* out_attn    = out_content + N * 256;
    unsigned int* vpk  = (unsigned int*)d_ws;          // 128*256 dwords = 128 KB

    pack_values_kernel<<<128, 256, 0, stream>>>(values, vpk);
    episodic_kernel<<<(int)(N / 128), 256, 0, stream>>>(
        query, W1, b1, W2, b2, keys, vpk, out_content, out_attn);
}
